// MultiHeadedAttention_9028021256866
// MI455X (gfx1250) — compile-verified, matrix operand order checked by reading, not run on hardware
//
#include <hip/hip_runtime.h>
#include <hip/hip_bf16.h>

typedef _Float16 half16 __attribute__((ext_vector_type(16)));
typedef _Float16 half8  __attribute__((ext_vector_type(8)));
typedef float    float8 __attribute__((ext_vector_type(8)));
typedef float    f32x4  __attribute__((ext_vector_type(4)));

#define D_MODEL 512
#define HEADS   8
#define DKDIM   64
#define SEQ     2048
#define BATCH   4

// Load a 16-element f16 A/B fragment: lane supplies base pointer already offset
// by its kb (0 or 8). Halves 0..7 = K..K+7, halves 8..15 = K+16..K+23 per the
// CDNA5 16-bit matrix VGPR layout (CDNA5 ISA §7.12.2).
__device__ __forceinline__ half16 load_frag16(const _Float16* p) {
  half8 lo = *(const half8*)p;
  half8 hi = *(const half8*)(p + 16);
  half16 f;
#pragma unroll
  for (int i = 0; i < 8; ++i) { f[i] = lo[i]; f[8 + i] = hi[i]; }
  return f;
}

__device__ __forceinline__ float8 wmma_f16(half16 a, half16 b, float8 c) {
  return __builtin_amdgcn_wmma_f32_16x16x32_f16(false, a, false, b, (short)0, c,
                                                false, false);
}

// ---- DPP16 cross-lane helpers: reduce within each 16-lane row group -------
// quad_perm(1,0,3,2)=0xB1 (xor1), quad_perm(2,3,0,1)=0x4E (xor2),
// row_mirror=0x140 (i->15-i), row_half_mirror=0x141 (i->7-i).
// The 4-step sequence fully reduces 16 lanes and never crosses the lane-16
// boundary, matching the WMMA C/D half-row layout. VALU-only: no LDS traffic.
template <int CTRL>
__device__ __forceinline__ float dpp_mov_f32(float x) {
  int v = __builtin_amdgcn_update_dpp(0, __builtin_bit_cast(int, x), CTRL, 0xf,
                                      0xf, true);
  return __builtin_bit_cast(float, v);
}
__device__ __forceinline__ float row16_max(float x) {
  x = fmaxf(x, dpp_mov_f32<0xB1>(x));
  x = fmaxf(x, dpp_mov_f32<0x4E>(x));
  x = fmaxf(x, dpp_mov_f32<0x140>(x));
  x = fmaxf(x, dpp_mov_f32<0x141>(x));
  return x;
}
__device__ __forceinline__ float row16_sum(float x) {
  x += dpp_mov_f32<0xB1>(x);
  x += dpp_mov_f32<0x4E>(x);
  x += dpp_mov_f32<0x140>(x);
  x += dpp_mov_f32<0x141>(x);
  return x;
}

// ---------------------------------------------------------------------------
// GEMM: y = x @ W^T + bias  (einsum 'bsd,ed->bse', W is [out_e, in_d] rowmajor)
// MODE 0/1/2: x is f32 [M,512]; output f16 scattered to [B,H,S,DK]
//             (MODE 0 additionally scales by 1/sqrt(DK) = 0.125 for Q).
// MODE 3    : x is f16 [M,512]; output f32 [M,512] to d_out.
// Block: 256 threads (8 waves). Tile: 128(M) x 64(N), K chunks of 32.
// Double-buffered LDS: next chunk staged into registers while WMMAs run on the
// current buffer; one barrier per chunk.
// ---------------------------------------------------------------------------
template <int MODE>
__global__ __launch_bounds__(256) void mha_gemm_wmma(
    const float* __restrict__ Xf, const _Float16* __restrict__ Xh,
    const float* __restrict__ W, const float* __restrict__ bias,
    _Float16* __restrict__ outH, float* __restrict__ outF) {
  __shared__ _Float16 As[2][128 * 32];  // A tile, f16, row stride 32
  __shared__ _Float16 Bs[2][64 * 32];   // Bs[n][k] = W[n0+n][k0+k]

  const int m0 = blockIdx.y * 128;
  const int n0 = blockIdx.x * 64;
  const int tid = threadIdx.x;
  const int wave = tid >> 5;
  const int lane = tid & 31;
  const int lh = lane & 15;
  const int kb = (lane >> 4) * 8;

  // per-thread staging coordinates
  const int ar = tid >> 1, ac = (tid & 1) * 16;   // A tile: 128 x 32
  const int br = tid >> 2, bc = (tid & 3) * 8;    // B tile:  64 x 32

  // ---- prologue: stage chunk 0 into buffer 0 ----
  {
    _Float16* adst = As[0] + ar * 32 + ac;
    if constexpr (MODE < 3) {
      const float* src = Xf + (size_t)(m0 + ar) * D_MODEL + ac;
#pragma unroll
      for (int i = 0; i < 4; ++i) {
        f32x4 d = *(const f32x4*)(src + i * 4);
#pragma unroll
        for (int j = 0; j < 4; ++j) adst[i * 4 + j] = (_Float16)d[j];
      }
    } else {
      const _Float16* src = Xh + (size_t)(m0 + ar) * D_MODEL + ac;
      *(half8*)adst = *(const half8*)src;
      *(half8*)(adst + 8) = *(const half8*)(src + 8);
    }
    const float* wsrc = W + (size_t)(n0 + br) * D_MODEL + bc;
    _Float16* bdst = Bs[0] + br * 32 + bc;
#pragma unroll
    for (int i = 0; i < 2; ++i) {
      f32x4 d = *(const f32x4*)(wsrc + i * 4);
#pragma unroll
      for (int j = 0; j < 4; ++j) bdst[i * 4 + j] = (_Float16)d[j];
    }
  }
  __syncthreads();

  float8 acc[4] = {};
  int buf = 0;
  for (int k0 = 0; k0 < D_MODEL; k0 += 32, buf ^= 1) {
    const bool have_next = (k0 + 32) < D_MODEL;
    // ---- issue global loads for next chunk early (latency hiding) ----
    f32x4 a32[4];  half8 a16[2];  f32x4 w32[2];
    if (have_next) {
      if constexpr (MODE < 3) {
        const float* src = Xf + (size_t)(m0 + ar) * D_MODEL + k0 + 32 + ac;
#pragma unroll
        for (int i = 0; i < 4; ++i) a32[i] = *(const f32x4*)(src + i * 4);
      } else {
        const _Float16* src = Xh + (size_t)(m0 + ar) * D_MODEL + k0 + 32 + ac;
        a16[0] = *(const half8*)src;
        a16[1] = *(const half8*)(src + 8);
      }
      const float* wsrc = W + (size_t)(n0 + br) * D_MODEL + k0 + 32 + bc;
#pragma unroll
      for (int i = 0; i < 2; ++i) w32[i] = *(const f32x4*)(wsrc + i * 4);
    }

    // ---- compute on current buffer ----
    const half16 afrag = load_frag16(As[buf] + (wave * 16 + lh) * 32 + kb);
#pragma unroll
    for (int t = 0; t < 4; ++t) {
      const half16 bfrag = load_frag16(Bs[buf] + (t * 16 + lh) * 32 + kb);
      acc[t] = wmma_f16(afrag, bfrag, acc[t]);
    }

    // ---- commit staged registers to the other buffer ----
    if (have_next) {
      _Float16* adst = As[buf ^ 1] + ar * 32 + ac;
      if constexpr (MODE < 3) {
#pragma unroll
        for (int i = 0; i < 4; ++i)
#pragma unroll
          for (int j = 0; j < 4; ++j) adst[i * 4 + j] = (_Float16)a32[i][j];
      } else {
        *(half8*)adst = a16[0];
        *(half8*)(adst + 8) = a16[1];
      }
      _Float16* bdst = Bs[buf ^ 1] + br * 32 + bc;
#pragma unroll
      for (int i = 0; i < 2; ++i)
#pragma unroll
        for (int j = 0; j < 4; ++j) bdst[i * 4 + j] = (_Float16)w32[i][j];
    }
    __syncthreads();
  }

  // ---- epilogue ----
  const int hs = lane >> 4;  // row-half select for C/D layout
#pragma unroll
  for (int t = 0; t < 4; ++t) {
    const int n = n0 + t * 16 + lh;
    const float bv = bias[n];
#pragma unroll
    for (int r = 0; r < 8; ++r) {
      const int m = m0 + wave * 16 + r + 8 * hs;
      float v = acc[t][r] + bv;
      if constexpr (MODE < 3) {
        if constexpr (MODE == 0) v *= 0.125f;  // 1/sqrt(DK), folded into Q
        const int b = m >> 11, s = m & 2047;
        const int h = n >> 6, dk = n & 63;
        outH[((((size_t)b * HEADS + h) * SEQ) + s) * DKDIM + dk] = (_Float16)v;
      } else {
        outF[(size_t)m * D_MODEL + n] = v;
      }
    }
  }
}

// ---------------------------------------------------------------------------
// Flash attention: grid (S/128 q-tiles, B*H). 256 threads = 8 waves; wave w
// owns 16 query rows. Streams 64-key blocks with double-buffered K/V tiles,
// software-pipelined mask loads (branch-free additive bias), and DPP16
// VALU-only softmax reductions: S=Q*K^T (8 wmma), online softmax, P via
// per-wave LDS D->A relayout, O += P*V (8 wmma).
// ---------------------------------------------------------------------------
__global__ __launch_bounds__(256) void mha_flash_attn_wmma(
    const _Float16* __restrict__ Qh, const _Float16* __restrict__ Kh,
    const _Float16* __restrict__ Vh, const int* __restrict__ mask,
    _Float16* __restrict__ Xh) {
  __shared__ _Float16 Ks[2][64 * 64];  // K tile natural: Ks[s][d]
  __shared__ _Float16 Vt[2][64 * 64];  // V tile transposed: Vt[d][s]
  __shared__ _Float16 Ps[8][16 * 64];  // per-wave P scratch (D->A relayout)

  const int bh = blockIdx.y;
  const int b = bh >> 3;
  const int h = bh & 7;
  const int q0 = blockIdx.x * 128;
  const int tid = threadIdx.x;
  const int wave = tid >> 5;
  const int lane = tid & 31;
  const int lh = lane & 15;
  const int kb = (lane >> 4) * 8;
  const int hs = lane >> 4;

  const _Float16* Qb = Qh + (size_t)bh * SEQ * DKDIM;
  const _Float16* Kb = Kh + (size_t)bh * SEQ * DKDIM;
  const _Float16* Vb = Vh + (size_t)bh * SEQ * DKDIM;
  const int* mrow = mask + b * SEQ;  // mask is [B,1,S]

  // staging coordinates: each thread owns one 64-wide row quarter (16 halves)
  const int sr = tid >> 2;
  const int sc2 = (tid & 3) * 16;

  // Q fragments are loop-invariant: preload straight from global (16B loads).
  half16 qfrag[2];
  {
    const _Float16* qp = Qb + (size_t)(q0 + wave * 16 + lh) * DKDIM;
    qfrag[0] = load_frag16(qp + kb);
    qfrag[1] = load_frag16(qp + 32 + kb);
  }

  // ---- prologue: stage key-block 0 into buffer 0; mask bias for block 0 ----
  float mb[4];
#pragma unroll
  for (int t = 0; t < 4; ++t)
    mb[t] = (mrow[t * 16 + lh] == 0) ? -1.0e9f : 0.0f;
  {
    const _Float16* ksrc = Kb + (size_t)sr * DKDIM + sc2;
    *(half8*)(Ks[0] + sr * 64 + sc2) = *(const half8*)ksrc;
    *(half8*)(Ks[0] + sr * 64 + sc2 + 8) = *(const half8*)(ksrc + 8);
    const _Float16* vsrc = Vb + (size_t)sr * DKDIM + sc2;
    half8 v0 = *(const half8*)vsrc;
    half8 v1 = *(const half8*)(vsrc + 8);
#pragma unroll
    for (int i = 0; i < 8; ++i) {
      Vt[0][(sc2 + i) * 64 + sr] = v0[i];
      Vt[0][(sc2 + 8 + i) * 64 + sr] = v1[i];
    }
  }
  __syncthreads();

  float8 o[4] = {};
  float mrun[8], lrun[8];
#pragma unroll
  for (int r = 0; r < 8; ++r) { mrun[r] = -1e30f; lrun[r] = 0.0f; }

  int buf = 0;
  for (int kk0 = 0; kk0 < SEQ; kk0 += 64, buf ^= 1) {
    const bool have_next = (kk0 + 64) < SEQ;
    // ---- issue next tile's global loads early (K, V, mask) ----
    half8 kr0, kr1, vr0, vr1;
    int mnext[4];
    if (have_next) {
      const _Float16* ksrc = Kb + (size_t)(kk0 + 64 + sr) * DKDIM + sc2;
      kr0 = *(const half8*)ksrc;
      kr1 = *(const half8*)(ksrc + 8);
      const _Float16* vsrc = Vb + (size_t)(kk0 + 64 + sr) * DKDIM + sc2;
      vr0 = *(const half8*)vsrc;
      vr1 = *(const half8*)(vsrc + 8);
#pragma unroll
      for (int t = 0; t < 4; ++t) mnext[t] = mrow[kk0 + 64 + t * 16 + lh];
    }

    // ---- S = Q * K^T : 4 key tiles x 2 d-chunks ----
    float8 sc[4] = {};
#pragma unroll
    for (int c = 0; c < 2; ++c) {
#pragma unroll
      for (int t = 0; t < 4; ++t) {
        const half16 kfrag =
            load_frag16(Ks[buf] + (t * 16 + lh) * 64 + c * 32 + kb);
        sc[t] = wmma_f16(qfrag[c], kfrag, sc[t]);
      }
    }
    // ---- mask: branch-free additive bias (pipelined load) ----
#pragma unroll
    for (int t = 0; t < 4; ++t)
#pragma unroll
      for (int r = 0; r < 8; ++r) sc[t][r] += mb[t];

    // ---- online softmax: row stats per (vgpr r, lane-half); DPP reduce ----
    float cfac[8];
#pragma unroll
    for (int r = 0; r < 8; ++r) {
      float rmax =
          fmaxf(fmaxf(sc[0][r], sc[1][r]), fmaxf(sc[2][r], sc[3][r]));
      rmax = row16_max(rmax);
      const float mn = fmaxf(mrun[r], rmax);
      cfac[r] = __expf(mrun[r] - mn);
      mrun[r] = mn;
    }
#pragma unroll
    for (int r = 0; r < 8; ++r) {
      float rsum = 0.0f;
#pragma unroll
      for (int t = 0; t < 4; ++t) {
        const float p = __expf(sc[t][r] - mrun[r]);
        sc[t][r] = p;
        rsum += p;
      }
      rsum = row16_sum(rsum);
      lrun[r] = lrun[r] * cfac[r] + rsum;
    }
#pragma unroll
    for (int t = 0; t < 4; ++t)
#pragma unroll
      for (int r = 0; r < 8; ++r) o[t][r] *= cfac[r];

    // ---- D-layout P -> A-layout via per-wave LDS (in-order, wait-free) ----
    _Float16* pp = Ps[wave];
#pragma unroll
    for (int t = 0; t < 4; ++t)
#pragma unroll
      for (int r = 0; r < 8; ++r)
        pp[(r + 8 * hs) * 64 + t * 16 + lh] = (_Float16)sc[t][r];

    // ---- O += P * V ----
#pragma unroll
    for (int c = 0; c < 2; ++c) {
      const half16 pfrag = load_frag16(pp + lh * 64 + c * 32 + kb);
#pragma unroll
      for (int t = 0; t < 4; ++t) {
        const half16 vfrag =
            load_frag16(Vt[buf] + (t * 16 + lh) * 64 + c * 32 + kb);
        o[t] = wmma_f16(pfrag, vfrag, o[t]);
      }
    }

    // ---- commit staged next tile to the other buffer; fold next mask ----
    if (have_next) {
      *(half8*)(Ks[buf ^ 1] + sr * 64 + sc2) = kr0;
      *(half8*)(Ks[buf ^ 1] + sr * 64 + sc2 + 8) = kr1;
#pragma unroll
      for (int i = 0; i < 8; ++i) {
        Vt[buf ^ 1][(sc2 + i) * 64 + sr] = vr0[i];
        Vt[buf ^ 1][(sc2 + 8 + i) * 64 + sr] = vr1[i];
      }
#pragma unroll
      for (int t = 0; t < 4; ++t)
        mb[t] = (mnext[t] == 0) ? -1.0e9f : 0.0f;
    }
    __syncthreads();
  }

  // ---- epilogue: X[b, q, h*64 + dk] = O / l ----
#pragma unroll
  for (int t = 0; t < 4; ++t) {
    const int n = t * 16 + lh;
#pragma unroll
    for (int r = 0; r < 8; ++r) {
      const int q = q0 + wave * 16 + r + 8 * hs;
      const float v = o[t][r] / lrun[r];
      Xh[((size_t)(b * SEQ + q)) * D_MODEL + h * DKDIM + n] = (_Float16)v;
    }
  }
}

// ---------------------------------------------------------------------------
extern "C" void kernel_launch(void* const* d_in, const int* in_sizes, int n_in,
                              void* d_out, int out_size, void* d_ws,
                              size_t ws_size, hipStream_t stream) {
  const float* q = (const float*)d_in[0];
  const float* k = (const float*)d_in[1];
  const float* v = (const float*)d_in[2];
  const int* mask = (const int*)d_in[3];
  const float* Wq = (const float*)d_in[4];
  const float* bq = (const float*)d_in[5];
  const float* Wk = (const float*)d_in[6];
  const float* bk = (const float*)d_in[7];
  const float* Wv = (const float*)d_in[8];
  const float* bv = (const float*)d_in[9];
  const float* Wo = (const float*)d_in[10];
  const float* bo = (const float*)d_in[11];
  float* out = (float*)d_out;

  const size_t elems = (size_t)BATCH * HEADS * SEQ * DKDIM;  // 4.19M halves
  _Float16* Qh = (_Float16*)d_ws;
  _Float16* Kh = Qh + elems;
  _Float16* Vh = Kh + elems;
  _Float16* Xh = Vh + elems;  // [B*S, 512] f16, 8MB; total ws use = 32MB

  const dim3 gemmGrid(D_MODEL / 64, (BATCH * SEQ) / 128);  // (8, 64)
  const dim3 blk(256);

  mha_gemm_wmma<0><<<gemmGrid, blk, 0, stream>>>(q, nullptr, Wq, bq, Qh, nullptr);
  mha_gemm_wmma<1><<<gemmGrid, blk, 0, stream>>>(k, nullptr, Wk, bk, Kh, nullptr);
  mha_gemm_wmma<2><<<gemmGrid, blk, 0, stream>>>(v, nullptr, Wv, bv, Vh, nullptr);

  mha_flash_attn_wmma<<<dim3(SEQ / 128, BATCH * HEADS), blk, 0, stream>>>(
      Qh, Kh, Vh, mask, Xh);

  mha_gemm_wmma<3><<<gemmGrid, blk, 0, stream>>>(nullptr, Xh, Wo, bo, nullptr,
                                                 out);
}
